// LengthBasedMLP_21758304321789
// MI455X (gfx1250) — matrix operand order repaired and checked by reading, not compiled, not run on hardware
//
#include <hip/hip_runtime.h>
#include <hip/hip_bf16.h>

// ---------------- problem constants ----------------
#define NDIV   8
#define NROWS  32768
#define IN_DIM 128
#define H1D    512
#define H2D    256
#define H3D    128
#define TILE_M 128
#define BLOCK_T 512
#define BN_EPS 1e-5f
#define NEG_SLOPE 0.01f

typedef _Float16 half16 __attribute__((ext_vector_type(16)));
typedef float    float8 __attribute__((ext_vector_type(8)));

__device__ __forceinline__ float8 wmma_f16(half16 a, half16 b, float8 c) {
  // D = A(16x32 f16) * B(32x16 f16) + C(16x16 f32)
  return __builtin_amdgcn_wmma_f32_16x16x32_f16(
      /*neg_a=*/false, a, /*neg_b=*/false, b,
      /*c_mod=*/(short)0, c, /*reuse_a=*/false, /*reuse_b=*/false);
}

// Build A operand from a global f32 row. Lane l<16: M=l, K={0..7,16..23}+kbase;
// lanes 16..31: M=l-16, K={8..15,24..31}+kbase. Caller passes xr = row + kbase + hi8.
__device__ __forceinline__ half16 load_a_f32(const float* __restrict__ xr) {
  half16 a;
#pragma unroll
  for (int j = 0; j < 8; ++j) {
    a[j]     = (_Float16)xr[j];
    a[j + 8] = (_Float16)xr[j + 16];
  }
  return a;
}

// Same but from f16 activations in LDS (two 16B chunks -> ds_load_b128 x2).
__device__ __forceinline__ half16 load_a_lds(const _Float16* p) {
  half16 a;
#pragma unroll
  for (int j = 0; j < 8; ++j) {
    a[j]     = p[j];
    a[j + 8] = p[j + 16];
  }
  return a;
}

// ---------------- kernel 1: fold BN into scale/shift, zero counters ----------------
__global__ void prep_scales(
    const float* __restrict__ b1, const float* __restrict__ g1, const float* __restrict__ be1,
    const float* __restrict__ m1, const float* __restrict__ v1,
    const float* __restrict__ b2, const float* __restrict__ g2, const float* __restrict__ be2,
    const float* __restrict__ m2, const float* __restrict__ v2,
    const float* __restrict__ b3, const float* __restrict__ g3, const float* __restrict__ be3,
    const float* __restrict__ m3, const float* __restrict__ v3,
    float* __restrict__ s1, float* __restrict__ c1,
    float* __restrict__ s2, float* __restrict__ c2,
    float* __restrict__ s3, float* __restrict__ c3,
    int* __restrict__ counts)
{
  const int i = blockIdx.x * blockDim.x + threadIdx.x;
  if (i < NDIV) counts[i] = 0;
  const int n1 = NDIV * H1D;            // 4096
  const int n2 = n1 + NDIV * H2D;       // 6144
  const int n3 = n2 + NDIV * H3D;       // 7168
  if (i < n1) {
    float sc = g1[i] * rsqrtf(v1[i] + BN_EPS);
    s1[i] = sc;
    c1[i] = (b1[i] - m1[i]) * sc + be1[i];
  } else if (i < n2) {
    int j = i - n1;
    float sc = g2[j] * rsqrtf(v2[j] + BN_EPS);
    s2[j] = sc;
    c2[j] = (b2[j] - m2[j]) * sc + be2[j];
  } else if (i < n3) {
    int j = i - n2;
    float sc = g3[j] * rsqrtf(v3[j] + BN_EPS);
    s3[j] = sc;
    c3[j] = (b3[j] - m3[j]) * sc + be3[j];
  }
}

// ---------------- kernel 2: pack weights f32 -> f16 in WMMA B-operand order ----------
// Packed layout (per expert, K x N): vector index r = (nt*KT + kt)*32 + lane,
// 16 contiguous halfs; element j <-> W[kt*32 + (lane>=16?8:0) + (j<8?j:j+8)][nt*16 + (lane&15)].
__global__ void pack_weights(
    const float* __restrict__ W1, const float* __restrict__ W2, const float* __restrict__ W3,
    _Float16* __restrict__ W1p, _Float16* __restrict__ W2p, _Float16* __restrict__ W3p)
{
  const int tid = blockIdx.x * blockDim.x + threadIdx.x;  // one 16-half vector each
  const int V1 = NDIV * IN_DIM * H1D / 16;  // 32768
  const int V2 = NDIV * H1D * H2D / 16;     // 65536
  const int V3 = NDIV * H2D * H3D / 16;     // 16384
  const float* src;
  _Float16* dst;
  int K, N, v;
  if (tid < V1)            { src = W1; dst = W1p; K = IN_DIM; N = H1D; v = tid; }
  else if (tid < V1 + V2)  { src = W2; dst = W2p; K = H1D;    N = H2D; v = tid - V1; }
  else if (tid < V1 + V2 + V3) { src = W3; dst = W3p; K = H2D; N = H3D; v = tid - V1 - V2; }
  else return;

  const int vecPerE = (K * N) / 16;
  const int e = v / vecPerE;
  const int r = v % vecPerE;
  const int lane = r & 31;
  const int g    = r >> 5;
  const int KT   = K / 32;
  const int nt   = g / KT;
  const int kt   = g % KT;
  const int n     = nt * 16 + (lane & 15);
  const int kbase = kt * 32 + ((lane >> 4) << 3);  // +8 for lanes 16..31
  const float* s = src + (size_t)e * K * N + (size_t)kbase * N + n;
  _Float16*   d = dst + (size_t)e * K * N + (size_t)r * 16;
#pragma unroll
  for (int j = 0; j < 16; ++j) d[j] = (_Float16)s[(size_t)(j + (j >= 8 ? 8 : 0)) * N];
}

// ---------------- kernel 3: routing (bucket + compaction) ----------------
__global__ void route_rows(const float* __restrict__ x, int* __restrict__ counts,
                           int* __restrict__ rowlist, float* __restrict__ out)
{
  const int row = blockIdx.x * blockDim.x + threadIdx.x;
  if (row >= NROWS) return;
  const float4* x4 = (const float4*)(x + (size_t)row * IN_DIM);
  float s = 0.f;
#pragma unroll
  for (int j = 0; j < IN_DIM / 4; ++j) {
    float4 p = x4[j];
    s += p.x + p.y + p.z + p.w;
  }
  // bucket i: i/D*L < s <= (i+1)/D*L  =>  i = ceil(s*D/L) - 1 ; D/L = 8/128
  const int b = (int)ceilf(s * (8.0f / 128.0f)) - 1;
  if (b >= 0 && b < NDIV) {
    int pos = atomicAdd(&counts[b], 1);
    rowlist[b * NROWS + pos] = row;
  } else {
    out[row] = 0.f;  // unrouted rows stay 0
  }
}

// ---------------- kernel 4: routed 4-layer MLP with WMMA ----------------
// 512 threads = 16 waves; wave w: M-subtile = w>>1 (16 rows, 0..7), N-half = w&1.
// 128 rows per block to cut L2 weight traffic (448 KB f16 weights re-read per block).
__global__ __launch_bounds__(BLOCK_T) void mlp_gemm(
    const float* __restrict__ x,
    const _Float16* __restrict__ W1p, const _Float16* __restrict__ W2p,
    const _Float16* __restrict__ W3p,
    const float* __restrict__ s1, const float* __restrict__ c1,
    const float* __restrict__ s2, const float* __restrict__ c2,
    const float* __restrict__ s3, const float* __restrict__ c3,
    const float* __restrict__ W4, const float* __restrict__ b4,
    const int* __restrict__ counts, const int* __restrict__ rowlist,
    float* __restrict__ out)
{
  __shared__ __align__(16) _Float16 sH1[TILE_M][H1D + 8];  // pitch 520, 133 KB
  __shared__ __align__(16) _Float16 sH2[TILE_M][H2D + 8];  // pitch 264, 67.6 KB
  __shared__ int sRows[TILE_M];

  const int e = blockIdx.y;
  const int cnt = counts[e];
  const int tile0 = blockIdx.x * TILE_M;
  if (tile0 >= cnt) return;
  const int nvalid = (cnt - tile0 < TILE_M) ? (cnt - tile0) : TILE_M;

  const int t = threadIdx.x;
  if (t < TILE_M) {
    const int idx = (t < nvalid) ? t : 0;  // pad with first row; masked at store
    sRows[t] = rowlist[e * NROWS + tile0 + idx];
  }
  __syncthreads();

  const int wid  = t >> 5;
  const int lane = t & 31;
  const int msub  = wid >> 1;           // M-subtile: 0..7 (16 rows each)
  const int nhalf = wid & 1;            // which half of the N tiles
  const int hi8   = ((lane >> 4) & 1) * 8;
  const int mloc  = msub * 16 + (lane & 15);  // local row for A operand

  const _Float16* w1e = W1p + (size_t)e * IN_DIM * H1D;
  const _Float16* w2e = W2p + (size_t)e * H1D * H2D;
  const _Float16* w3e = W3p + (size_t)e * H2D * H3D;

  // ---------- Layer 1: [16x128] @ [128x512], A cached in registers ----------
  {
    const int row = sRows[mloc];
    const float* xrow = x + (size_t)row * IN_DIM;
    half16 a[4];
#pragma unroll
    for (int kt = 0; kt < 4; ++kt) a[kt] = load_a_f32(xrow + kt * 32 + hi8);

    const float* s1e = s1 + e * H1D;
    const float* c1e = c1 + e * H1D;
    for (int i = 0; i < 16; ++i) {
      const int nt = nhalf * 16 + i;
      float8 acc = {0.f, 0.f, 0.f, 0.f, 0.f, 0.f, 0.f, 0.f};
#pragma unroll
      for (int kt = 0; kt < 4; ++kt) {
        half16 b = *(const half16*)(w1e + (((size_t)(nt * 4 + kt)) * 32 + lane) * 16);
        acc = wmma_f16(a[kt], b, acc);
      }
      const int n = nt * 16 + (lane & 15);
      const float sc = s1e[n], sh = c1e[n];
#pragma unroll
      for (int r = 0; r < 8; ++r) {
        float v = acc[r] * sc + sh;
        v = (v >= 0.f) ? v : NEG_SLOPE * v;
        sH1[msub * 16 + hi8 + r][n] = (_Float16)v;
      }
    }
  }
  __syncthreads();

  // ---------- Layer 2: [16x512] @ [512x256], K-blocked A reuse ----------
  {
    const float* s2e = s2 + e * H2D;
    const float* c2e = c2 + e * H2D;
    float8 acc[8];
#pragma unroll
    for (int i = 0; i < 8; ++i) acc[i] = (float8){0.f,0.f,0.f,0.f,0.f,0.f,0.f,0.f};
    for (int ktb = 0; ktb < 4; ++ktb) {
      half16 a[4];
#pragma unroll
      for (int kk = 0; kk < 4; ++kk)
        a[kk] = load_a_lds(&sH1[mloc][(ktb * 4 + kk) * 32 + hi8]);
#pragma unroll
      for (int i = 0; i < 8; ++i) {
        const int nt = nhalf * 8 + i;
#pragma unroll
        for (int kk = 0; kk < 4; ++kk) {
          const int kt = ktb * 4 + kk;
          half16 b = *(const half16*)(w2e + (((size_t)(nt * 16 + kt)) * 32 + lane) * 16);
          acc[i] = wmma_f16(a[kk], b, acc[i]);
        }
      }
    }
#pragma unroll
    for (int i = 0; i < 8; ++i) {
      const int nt = nhalf * 8 + i;
      const int n = nt * 16 + (lane & 15);
      const float sc = s2e[n], sh = c2e[n];
#pragma unroll
      for (int r = 0; r < 8; ++r) {
        float v = acc[i][r] * sc + sh;
        v = (v >= 0.f) ? v : NEG_SLOPE * v;
        sH2[msub * 16 + hi8 + r][n] = (_Float16)v;
      }
    }
  }
  __syncthreads();

  // ---------- Layer 3: [16x256] @ [256x128] -> h3 reuses sH1 (pitch 136) ----------
  _Float16* sH3 = &sH1[0][0];
  {
    const float* s3e = s3 + e * H3D;
    const float* c3e = c3 + e * H3D;
    float8 acc[4];
#pragma unroll
    for (int i = 0; i < 4; ++i) acc[i] = (float8){0.f,0.f,0.f,0.f,0.f,0.f,0.f,0.f};
    for (int ktb = 0; ktb < 2; ++ktb) {
      half16 a[4];
#pragma unroll
      for (int kk = 0; kk < 4; ++kk)
        a[kk] = load_a_lds(&sH2[mloc][(ktb * 4 + kk) * 32 + hi8]);
#pragma unroll
      for (int i = 0; i < 4; ++i) {
        const int nt = nhalf * 4 + i;
#pragma unroll
        for (int kk = 0; kk < 4; ++kk) {
          const int kt = ktb * 4 + kk;
          half16 b = *(const half16*)(w3e + (((size_t)(nt * 8 + kt)) * 32 + lane) * 16);
          acc[i] = wmma_f16(a[kk], b, acc[i]);
        }
      }
    }
#pragma unroll
    for (int i = 0; i < 4; ++i) {
      const int nt = nhalf * 4 + i;
      const int n = nt * 16 + (lane & 15);
      const float sc = s3e[n], sh = c3e[n];
#pragma unroll
      for (int r = 0; r < 8; ++r) {
        float v = acc[i][r] * sc + sh;
        v = (v >= 0.f) ? v : NEG_SLOPE * v;
        sH3[(size_t)(msub * 16 + hi8 + r) * 136 + n] = (_Float16)v;
      }
    }
  }
  __syncthreads();

  // ---------- Layer 4: [rows x 128] . W4 + b4 (VALU dot) ----------
  if (t < nvalid) {
    const float* w4e = W4 + e * H3D;
    float acc = b4[e];
    const _Float16* h = sH3 + (size_t)t * 136;
#pragma unroll 8
    for (int j = 0; j < H3D; ++j) acc += (float)h[j] * w4e[j];
    out[sRows[t]] = acc;
  }
}

// ---------------- host launcher ----------------
extern "C" void kernel_launch(void* const* d_in, const int* in_sizes, int n_in,
                              void* d_out, int out_size, void* d_ws, size_t ws_size,
                              hipStream_t stream) {
  (void)in_sizes; (void)n_in; (void)out_size; (void)ws_size;

  const float* x   = (const float*)d_in[0];
  const float* W1  = (const float*)d_in[1];
  const float* b1  = (const float*)d_in[2];
  const float* g1  = (const float*)d_in[3];
  const float* be1 = (const float*)d_in[4];
  const float* m1  = (const float*)d_in[5];
  const float* v1  = (const float*)d_in[6];
  const float* W2  = (const float*)d_in[7];
  const float* b2  = (const float*)d_in[8];
  const float* g2  = (const float*)d_in[9];
  const float* be2 = (const float*)d_in[10];
  const float* m2  = (const float*)d_in[11];
  const float* v2  = (const float*)d_in[12];
  const float* W3  = (const float*)d_in[13];
  const float* b3  = (const float*)d_in[14];
  const float* g3  = (const float*)d_in[15];
  const float* be3 = (const float*)d_in[16];
  const float* m3  = (const float*)d_in[17];
  const float* v3  = (const float*)d_in[18];
  const float* W4  = (const float*)d_in[19];
  const float* b4  = (const float*)d_in[20];
  float* out = (float*)d_out;

  // workspace carve-up (bytes)
  char* ws = (char*)d_ws;
  int*       counts  = (int*)(ws + 0);                       // 8 ints
  int*       rowlist = (int*)(ws + 128);                     // 8*32768*4 = 1,048,576
  _Float16*  W1p     = (_Float16*)(ws + 1048704);            // 1,048,576 B
  _Float16*  W2p     = (_Float16*)(ws + 2097280);            // 2,097,152 B
  _Float16*  W3p     = (_Float16*)(ws + 4194432);            //   524,288 B
  float*     s1      = (float*)(ws + 4718720);               // 16,384 B
  float*     c1      = (float*)(ws + 4735104);
  float*     s2      = (float*)(ws + 4751488);               //  8,192 B
  float*     c2      = (float*)(ws + 4759680);
  float*     s3      = (float*)(ws + 4767872);               //  4,096 B
  float*     c3      = (float*)(ws + 4771968);

  // 1) BN folding + zero counters (7168 threads)
  prep_scales<<<28, 256, 0, stream>>>(b1, g1, be1, m1, v1,
                                      b2, g2, be2, m2, v2,
                                      b3, g3, be3, m3, v3,
                                      s1, c1, s2, c2, s3, c3, counts);

  // 2) weight pack f32 -> f16 WMMA-B layout (114688 vectors)
  pack_weights<<<448, 256, 0, stream>>>(W1, W2, W3, W1p, W2p, W3p);

  // 3) routing / compaction
  route_rows<<<NROWS / 256, 256, 0, stream>>>(x, counts, rowlist, out);

  // 4) routed expert MLP (worst case: all rows in one expert)
  dim3 grid(NROWS / TILE_M, NDIV);
  mlp_gemm<<<grid, BLOCK_T, 0, stream>>>(x, W1p, W2p, W3p,
                                         s1, c1, s2, c2, s3, c3,
                                         W4, b4, counts, rowlist, out);
}
